// MoELoRA_3805341024604
// MI455X (gfx1250) — compile-verified, matrix operand order checked by reading, not run on hardware
//
#include <hip/hip_runtime.h>
#include <stdint.h>

typedef __attribute__((ext_vector_type(16))) __bf16 v16bf;
typedef __attribute__((ext_vector_type(8)))  __bf16 v8bf;
typedef __attribute__((ext_vector_type(8)))  float  v8f;
typedef int v4i __attribute__((vector_size(16)));   // matches builtin param type

#define BSZ 16
#define SEQ 512
#define IN_DIM 1024
#define OUT_DIM 1024
#define RANK 16
#define TOPK 4
#define KR 64   // TOPK*RANK
#define NKSTEP (IN_DIM / 32)

__device__ __forceinline__ uint16_t f2bfu(float f) {
    uint32_t u = __builtin_bit_cast(uint32_t, f);
    u += 0x7FFFu + ((u >> 16) & 1u);     // round-to-nearest-even
    return (uint16_t)(u >> 16);
}
__device__ __forceinline__ __bf16 f2bf(float f) {
    uint16_t h = f2bfu(f);
    return __builtin_bit_cast(__bf16, h);
}
__device__ __forceinline__ v16bf make_frag(uint4 a0, uint4 a1) {
    v8bf lo = __builtin_bit_cast(v8bf, a0);
    v8bf hi = __builtin_bit_cast(v8bf, a1);
    return __builtin_shufflevector(lo, hi, 0,1,2,3,4,5,6,7,8,9,10,11,12,13,14,15);
}

// --- CDNA5 async global -> LDS copy (ASYNCcnt-tracked, no VGPR data staging) ---
__device__ __forceinline__ void async_ld_b128(const uint16_t* g, uint16_t* l) {
#if __has_builtin(__builtin_amdgcn_global_load_async_to_lds_b128)
    __builtin_amdgcn_global_load_async_to_lds_b128(
        (__attribute__((address_space(1))) v4i*)(uintptr_t)g,
        (__attribute__((address_space(3))) v4i*)(uint32_t)(uintptr_t)l,
        0, 0);
#else
    uint32_t lo = (uint32_t)(uintptr_t)l;   // LDS byte address (aperture low bits)
    asm volatile("global_load_async_to_lds_b128 %0, %1, off"
                 :: "v"(lo), "v"(g) : "memory");
#endif
}
#if __has_builtin(__builtin_amdgcn_s_wait_asynccnt)
#define WAIT_ASYNC(n) __builtin_amdgcn_s_wait_asynccnt(n)
#else
#define WAIT_ASYNC(n) asm volatile("s_wait_asynccnt %0" :: "n"(n) : "memory")
#endif

// ---------------- fp32 -> bf16 bulk conversion (8 elems / thread) ----------------
__global__ void k_cvt_bf16(const float* __restrict__ src, uint16_t* __restrict__ dst, int n8) {
    int i = blockIdx.x * blockDim.x + threadIdx.x;
    if (i >= n8) return;
    const float4* s = (const float4*)src + (size_t)i * 2;
    float4 f0 = s[0], f1 = s[1];
    v8bf o;
    o[0] = f2bf(f0.x); o[1] = f2bf(f0.y); o[2] = f2bf(f0.z); o[3] = f2bf(f0.w);
    o[4] = f2bf(f1.x); o[5] = f2bf(f1.y); o[6] = f2bf(f1.z); o[7] = f2bf(f1.w);
    *((v8bf*)dst + i) = o;
}

// ------------- per-batch expert gather: AeffT [b][64][1024], BeffT [b][1024][64],
// ------------- bias_total [b][1024] (attn scaling folded into Beff & bias) -------
__global__ void k_prep(const float* __restrict__ A_pool, const float* __restrict__ B_pool,
                       const float* __restrict__ bias_pool, const float* __restrict__ bias,
                       const float* __restrict__ attn, const int* __restrict__ idx,
                       uint16_t* __restrict__ AeffT, uint16_t* __restrict__ BeffT,
                       float* __restrict__ bias_total) {
    const int b = blockIdx.x, tid = threadIdx.x;
    int   e[TOPK]; float w[TOPK];
    #pragma unroll
    for (int k = 0; k < TOPK; k++) { e[k] = idx[b * TOPK + k]; w[k] = attn[b * TOPK + k]; }

    // AeffT[b][j][i] = A_pool[e_j][i][r_j]  (zeroed if attn==0, matching masked_fill)
    for (int t = tid; t < KR * IN_DIM; t += blockDim.x) {
        int j = t >> 10, i = t & (IN_DIM - 1);
        float v = A_pool[((size_t)e[j >> 4] * IN_DIM + i) * RANK + (j & 15)];
        if (w[j >> 4] == 0.0f) v = 0.0f;
        AeffT[((size_t)b * KR + j) * IN_DIM + i] = f2bfu(v);
    }
    // BeffT[b][o][j] = attn[b,k_j] * B_pool[e_j][r_j][o]
    for (int t = tid; t < OUT_DIM * KR; t += blockDim.x) {
        int o = t >> 6, j = t & 63;
        float v = w[j >> 4] * B_pool[((size_t)e[j >> 4] * RANK + (j & 15)) * OUT_DIM + o];
        BeffT[((size_t)b * OUT_DIM + o) * KR + j] = f2bfu(v);
    }
    // bias_total[b][o] = bias[o] + sum_k attn*bias_pool[e_k][o]   (SCALE == 1)
    for (int o = tid; o < OUT_DIM; o += blockDim.x) {
        float s = bias[o];
        #pragma unroll
        for (int k = 0; k < TOPK; k++) s += w[k] * bias_pool[(size_t)e[k] * OUT_DIM + o];
        bias_total[(size_t)b * OUT_DIM + o] = s;
    }
}

// ---------------- T[b] = x_bf16[b] @ Aeff[b]  ->  T_bf16 [b][512][64] ----------------
__global__ void __launch_bounds__(256) k_tgemm(const uint16_t* __restrict__ xb,
                                               const uint16_t* __restrict__ AeffT,
                                               uint16_t* __restrict__ Tb) {
    const int b = blockIdx.x >> 2, mblk = blockIdx.x & 3;
    const int wave = threadIdx.x >> 5, lane = threadIdx.x & 31;
    const int half = lane >> 4, ln = lane & 15;
    const int rb = mblk * 128 + wave * 16;          // 16 rows per wave
    const v8f vzero = {0.f,0.f,0.f,0.f,0.f,0.f,0.f,0.f};
    v8f c[4];
    #pragma unroll
    for (int ni = 0; ni < 4; ni++) c[ni] = vzero;

    for (int kk = 0; kk < NKSTEP; kk++) {
        const int k0 = kk * 32;
        const uint16_t* ap = xb + ((size_t)(b * SEQ) + rb + ln) * IN_DIM + k0;
        v16bf af = make_frag(*(const uint4*)(ap + 8 * half),
                             *(const uint4*)(ap + 16 + 8 * half));
        #pragma unroll
        for (int ni = 0; ni < 4; ni++) {
            const uint16_t* bp = AeffT + ((size_t)b * KR + 16 * ni + ln) * IN_DIM + k0;
            v16bf bf = make_frag(*(const uint4*)(bp + 8 * half),
                                 *(const uint4*)(bp + 16 + 8 * half));
            c[ni] = __builtin_amdgcn_wmma_f32_16x16x32_bf16(false, af, false, bf,
                                                            (short)0, c[ni], false, false);
        }
    }
    #pragma unroll
    for (int ni = 0; ni < 4; ni++)
        #pragma unroll
        for (int v = 0; v < 8; v++)
            Tb[((size_t)(b * SEQ) + rb + 8 * half + v) * KR + 16 * ni + ln] = f2bfu(c[ni][v]);
}

// --------- main fused GEMM: out = x@W^T (K=1024) + T@Beff (K=64) + bias_total ---------
// 128x128 block tile, 8 waves as 4(M) x 2(N), 32x64 per wave (2x4 WMMA accum tiles),
// double-buffered LDS filled by async global->LDS copies.
#define LDSR 40   // 32 bf16 + 8 pad -> 80B row stride, conflict-free ds_load_b128
__global__ void __launch_bounds__(256) k_main(const uint16_t* __restrict__ xb,
                                              const uint16_t* __restrict__ wb,
                                              const uint16_t* __restrict__ Tb,
                                              const uint16_t* __restrict__ BeffT,
                                              const float* __restrict__ bias_total,
                                              float* __restrict__ out) {
    const int obase = blockIdx.x * 128, nbase = blockIdx.y * 128, b = blockIdx.z;
    const int tid = threadIdx.x;
    const int wave = tid >> 5, lane = tid & 31, half = lane >> 4, ln = lane & 15;
    const int wm = wave >> 1, wn = wave & 1;      // 4 x 2 wave grid, 32x64 per wave

    __shared__ __align__(16) uint16_t Xs[2][128 * LDSR];
    __shared__ __align__(16) uint16_t Ws[2][128 * LDSR];

    const v8f vzero = {0.f,0.f,0.f,0.f,0.f,0.f,0.f,0.f};
    v8f acc[2][4];
    #pragma unroll
    for (int mi = 0; mi < 2; mi++)
        #pragma unroll
        for (int ni = 0; ni < 4; ni++) acc[mi][ni] = vzero;

    // staging: 256 threads cover 128 rows x 4 x 16B chunks in two row-halves
    const int srow = tid >> 2, schunk = (tid & 3) * 8;
    const uint16_t* xg0 = xb + ((size_t)(b * SEQ) + nbase + srow) * IN_DIM + schunk;
    const uint16_t* xg1 = xg0 + (size_t)64 * IN_DIM;
    const uint16_t* wg0 = wb + ((size_t)(obase + srow)) * IN_DIM + schunk;
    const uint16_t* wg1 = wg0 + (size_t)64 * IN_DIM;
    const int l0 = srow * LDSR + schunk, l1 = (64 + srow) * LDSR + schunk;

    // prologue: tile 0 in flight
    async_ld_b128(xg0, &Xs[0][l0]);
    async_ld_b128(xg1, &Xs[0][l1]);
    async_ld_b128(wg0, &Ws[0][l0]);
    async_ld_b128(wg1, &Ws[0][l1]);

    for (int kk = 0; kk < NKSTEP; kk++) {
        const int cur = kk & 1;
        if (kk + 1 < NKSTEP) {               // issue next K-slab into the other buffer
            const int k1 = (kk + 1) * 32;
            async_ld_b128(xg0 + k1, &Xs[cur ^ 1][l0]);
            async_ld_b128(xg1 + k1, &Xs[cur ^ 1][l1]);
            async_ld_b128(wg0 + k1, &Ws[cur ^ 1][l0]);
            async_ld_b128(wg1 + k1, &Ws[cur ^ 1][l1]);
            WAIT_ASYNC(4);                   // tile kk landed; 4 next-tile loads in flight
        } else {
            WAIT_ASYNC(0);
        }
        __syncthreads();                     // tile kk visible to all waves

        v16bf a[2], bf[4];
        #pragma unroll
        for (int mi = 0; mi < 2; mi++) {
            const int r = (32 * wm + 16 * mi + ln) * LDSR;
            a[mi] = make_frag(*(const uint4*)&Xs[cur][r + 8 * half],
                              *(const uint4*)&Xs[cur][r + 16 + 8 * half]);
        }
        #pragma unroll
        for (int ni = 0; ni < 4; ni++) {
            const int r = (64 * wn + 16 * ni + ln) * LDSR;
            bf[ni] = make_frag(*(const uint4*)&Ws[cur][r + 8 * half],
                               *(const uint4*)&Ws[cur][r + 16 + 8 * half]);
        }
        #pragma unroll
        for (int mi = 0; mi < 2; mi++)
            #pragma unroll
            for (int ni = 0; ni < 4; ni++)
                acc[mi][ni] = __builtin_amdgcn_wmma_f32_16x16x32_bf16(
                    false, a[mi], false, bf[ni], (short)0, acc[mi][ni], false, false);

        __syncthreads();                     // all reads of buf[cur] done before reuse
    }

    // LoRA rank-64 accumulation: acc += T[b] @ Beff[b] (2 K-steps, direct global frags)
    #pragma unroll
    for (int kk = 0; kk < 2; kk++) {
        const int k0 = kk * 32;
        v16bf a2[2], b2[4];
        #pragma unroll
        for (int mi = 0; mi < 2; mi++) {
            const int row = nbase + 32 * wm + 16 * mi + ln;
            const uint16_t* tp = Tb + ((size_t)(b * SEQ) + row) * KR + k0;
            a2[mi] = make_frag(*(const uint4*)(tp + 8 * half),
                               *(const uint4*)(tp + 16 + 8 * half));
        }
        #pragma unroll
        for (int ni = 0; ni < 4; ni++) {
            const int col = obase + 64 * wn + 16 * ni + ln;
            const uint16_t* bp = BeffT + ((size_t)(b * OUT_DIM) + col) * KR + k0;
            b2[ni] = make_frag(*(const uint4*)(bp + 8 * half),
                               *(const uint4*)(bp + 16 + 8 * half));
        }
        #pragma unroll
        for (int mi = 0; mi < 2; mi++)
            #pragma unroll
            for (int ni = 0; ni < 4; ni++)
                acc[mi][ni] = __builtin_amdgcn_wmma_f32_16x16x32_bf16(
                    false, a2[mi], false, b2[ni], (short)0, acc[mi][ni], false, false);
    }

    // epilogue: + bias_total, fp32 store.  C/D layout: row = vgpr + 8*half, col = ln
    #pragma unroll
    for (int ni = 0; ni < 4; ni++) {
        const int col = obase + 64 * wn + 16 * ni + ln;
        const float bv = bias_total[(size_t)b * OUT_DIM + col];
        #pragma unroll
        for (int mi = 0; mi < 2; mi++) {
            const int row0 = nbase + 32 * wm + 16 * mi + 8 * half;
            #pragma unroll
            for (int v = 0; v < 8; v++)
                out[((size_t)(b * SEQ) + row0 + v) * OUT_DIM + col] = acc[mi][ni][v] + bv;
        }
    }
}

extern "C" void kernel_launch(void* const* d_in, const int* in_sizes, int n_in,
                              void* d_out, int out_size, void* d_ws, size_t ws_size,
                              hipStream_t stream) {
    const float* x         = (const float*)d_in[0];
    const float* weight    = (const float*)d_in[1];
    const float* bias      = (const float*)d_in[2];
    const float* A_pool    = (const float*)d_in[3];
    const float* B_pool    = (const float*)d_in[4];
    const float* bias_pool = (const float*)d_in[5];
    const float* attn      = (const float*)d_in[6];
    const int*   idx       = (const int*)d_in[7];
    // d_in[8] frozen_mask: forward-identical (stop_gradient only) -> unused
    float* out = (float*)d_out;

    uint8_t* ws = (uint8_t*)d_ws;
    size_t off = 0;
    uint16_t* xb    = (uint16_t*)(ws + off); off += (size_t)BSZ * SEQ * IN_DIM * 2;   // 16 MB
    uint16_t* wb    = (uint16_t*)(ws + off); off += (size_t)OUT_DIM * IN_DIM * 2;     //  2 MB
    uint16_t* AeffT = (uint16_t*)(ws + off); off += (size_t)BSZ * KR * IN_DIM * 2;    //  2 MB
    uint16_t* BeffT = (uint16_t*)(ws + off); off += (size_t)BSZ * OUT_DIM * KR * 2;   //  2 MB
    uint16_t* Tb    = (uint16_t*)(ws + off); off += (size_t)BSZ * SEQ * KR * 2;       //  1 MB
    float* bias_tot = (float*)(ws + off);    off += (size_t)BSZ * OUT_DIM * 4;

    const int nx8 = BSZ * SEQ * IN_DIM / 8;
    const int nw8 = OUT_DIM * IN_DIM / 8;
    k_cvt_bf16<<<(nx8 + 255) / 256, 256, 0, stream>>>(x, xb, nx8);
    k_cvt_bf16<<<(nw8 + 255) / 256, 256, 0, stream>>>(weight, wb, nw8);
    k_prep<<<BSZ, 256, 0, stream>>>(A_pool, B_pool, bias_pool, bias, attn, idx,
                                    AeffT, BeffT, bias_tot);
    k_tgemm<<<BSZ * 4, 256, 0, stream>>>(xb, AeffT, Tb);
    k_main<<<dim3(OUT_DIM / 128, SEQ / 128, BSZ), 256, 0, stream>>>(xb, wb, Tb, BeffT,
                                                                    bias_tot, out);
}
